// GraphUNetDualTransformer_42099269435634
// MI455X (gfx1250) — matrix operand order repaired and checked by reading, not compiled, not run on hardware
//
#include <hip/hip_runtime.h>
#include <math.h>

// ---------------------------------------------------------------------------
// Problem constants (from reference)
// ---------------------------------------------------------------------------
#define Bc   2
#define Nn   1024
#define Ff   8
#define Hh   64
#define Tt   16
#define ER   8192
#define EC   32768
#define NFc  (Nn*Ff)          // 8192
#define MC   (Bc*Tt*NFc)      // 262144  causal rows
#define MR   (Bc*Tt*Nn)       // 32768   river rows
#define BNT  (Bc*Nn*Tt)       // 32768   fusion rows
#define XOUT_ELEMS (Bc*Nn*Hh*Tt)  // 2097152

// ln(10000)/H
#define PE_SCALE 0.14391157f

typedef __attribute__((ext_vector_type(16))) __bf16 bf16x16;
typedef __attribute__((ext_vector_type(8)))  __bf16 bf16x8;
typedef __attribute__((ext_vector_type(8)))  float  f32x8;

__device__ __forceinline__ __bf16 f2bf(float x) { return (__bf16)x; }

// ---------------------------------------------------------------------------
// WMMA GEMM:  C[M,64] = act( A[M,K] @ W[K,64] + bias )
//   A is bf16 row-major; K<=128 concat mode: k<SPLIT from A0, else A1.
//   W supplied as f32 [K,64] row-major; converted to bf16 column-major in LDS.
//   256 threads = 8 waves; each wave: 16 rows x 64 cols via 4 accumulators.
//   ACT: 0=none 1=leaky_relu 2=silu
//   OUTM: 0=f32 [M,64], 1=bf16 [M,64], 2=f32 transposed x_out[(m>>4)*1024+c*16+(m&15)]
// ---------------------------------------------------------------------------
union AFrag { bf16x16 v; bf16x8 h[2]; };
union BFrag { bf16x16 v; bf16x8 h[2]; };

template<int K, int SPLIT, int ACT, int OUTM>
__global__ __launch_bounds__(256) void gemm_wmma(
    const __bf16* __restrict__ A0, int lda0,
    const __bf16* __restrict__ A1, int lda1,
    const float*  __restrict__ Wsrc,   // [K,64] f32 row-major
    const float*  __restrict__ bias,   // [64]
    void* __restrict__ Cout)
{
    // Stage W into LDS, bf16, column-major: w_lds[col*K + k]
    __shared__ __bf16 w_lds[64 * K];
    for (int i = threadIdx.x; i < K * 64; i += 256) {
        int k = i >> 6, c = i & 63;
        w_lds[c * K + k] = f2bf(Wsrc[i]);
    }
    __syncthreads();

    const int lane = threadIdx.x & 31;
    const int wave = threadIdx.x >> 5;
    const int half = lane >> 4;      // 0: lanes 0-15, 1: lanes 16-31
    const int l16  = lane & 15;
    const long r0  = (long)blockIdx.x * 128 + (long)wave * 16;

    f32x8 acc[4];
#pragma unroll
    for (int nt = 0; nt < 4; ++nt) {
        float bv = bias[nt * 16 + l16];
#pragma unroll
        for (int i = 0; i < 8; ++i) acc[nt][i] = bv;
    }

#pragma unroll
    for (int kk = 0; kk < K; kk += 32) {
        // A fragment, ISA 16-bit A 16x32 layout:
        // lane<16 : K kk+{0..7}, kk+{16..23}; lane>=16 : K kk+{8..15}, kk+{24..31}
        AFrag a;
        const __bf16* base;
        if (kk < SPLIT) base = A0 + (r0 + l16) * (long)lda0 + kk;
        else            base = A1 + (r0 + l16) * (long)lda1 + (kk - SPLIT);
        a.h[0] = *(const bf16x8*)(base + half * 8);
        a.h[1] = *(const bf16x8*)(base + 16 + half * 8);

#pragma unroll
        for (int nt = 0; nt < 4; ++nt) {
            // B fragment, 16-bit B 32x16: lane<16 -> col, K kk+{0..7}, kk+{16..23} ;
            // lane>=16 -> col, K kk+{8..15}, kk+{24..31}  (column-major LDS)
            const int col = nt * 16 + l16;
            const __bf16* wp = &w_lds[col * K + kk + half * 8];
            BFrag b;
            b.h[0] = *(const bf16x8*)(wp);
            b.h[1] = *(const bf16x8*)(wp + 16);
            acc[nt] = __builtin_amdgcn_wmma_f32_16x16x32_bf16(
                false, a.v, false, b.v, (short)0, acc[nt], false, false);
        }
    }

#pragma unroll
    for (int nt = 0; nt < 4; ++nt) {
        const int col = nt * 16 + l16;
#pragma unroll
        for (int i = 0; i < 8; ++i) {
            long row = r0 + half * 8 + i;   // C/D layout: vgpr i, half selects +8
            float v = acc[nt][i];
            if (ACT == 1)      v = v >= 0.f ? v : 0.01f * v;
            else if (ACT == 2) v = v / (1.f + __expf(-v));
            if (OUTM == 0)      ((float*)Cout)[row * 64 + col] = v;
            else if (OUTM == 1) ((__bf16*)Cout)[row * 64 + col] = f2bf(v);
            else ((float*)Cout)[(row >> 4) * 1024 + (long)col * 16 + (row & 15)] = v;
        }
    }
}

// ---------------------------------------------------------------------------
// Elementwise / prep kernels
// ---------------------------------------------------------------------------
__global__ void k_zero_f32(float* p, long n) {
    long i = (long)blockIdx.x * 256 + threadIdx.x;
    if (i < n) p[i] = 0.f;
}

__global__ void k_f32_to_bf16(const float* __restrict__ s, __bf16* __restrict__ d, long n) {
    long i = (long)blockIdx.x * 256 + threadIdx.x;
    if (i < n) d[i] = f2bf(s[i]);
}

// river edge weights: mw = clip(sigmoid(gate) * (attr@eW + eb), 0, 1)
__global__ void k_edge_w_river(const float* __restrict__ attr, const float* __restrict__ eW,
                               const float* __restrict__ eb, const float* __restrict__ gate,
                               float* __restrict__ mw) {
    int e = blockIdx.x * 256 + threadIdx.x;
    if (e >= ER) return;
    float ew = eb[0];
#pragma unroll
    for (int a = 0; a < 4; ++a) ew += attr[e * 4 + a] * eW[a];
    float g = 1.f / (1.f + __expf(-gate[0]));
    mw[e] = fminf(fmaxf(g * ew, 0.f), 1.f);
}

__global__ void k_edge_w_causal(const float* __restrict__ cw, const float* __restrict__ gate,
                                float* __restrict__ mw) {
    int e = blockIdx.x * 256 + threadIdx.x;
    if (e >= EC) return;
    float g = 1.f / (1.f + __expf(-gate[0]));
    mw[e] = fminf(fmaxf(g * cw[e], 0.f), 1.f);
}

// nodes[(bt*NF + n*F+f)*64 + h] = pfx[b,n,f,h,t] + femb[f,h]   (bf16)
__global__ void k_prep_nodes(const float* __restrict__ pfx, const float* __restrict__ femb,
                             __bf16* __restrict__ out) {
    long gid = (long)blockIdx.x * 256 + threadIdx.x;
    if (gid >= (long)MC * 64) return;
    int h = gid & 63;
    long m = gid >> 6;
    int f = m & 7;
    int n = (int)((m >> 3) & (Nn - 1));
    long bt = m >> 13;
    int t = (int)(bt & (Tt - 1));
    int b = (int)(bt >> 4);
    long src = ((((long)b * Nn + n) * Ff + f) * Hh + h) * Tt + t;
    out[gid] = f2bf(pfx[src] + femb[f * Hh + h]);
}

// xbt[(bt*N + n)*64 + h] = x_global[b,n,h,t]   (bf16)
__global__ void k_prep_xbt(const float* __restrict__ xg, __bf16* __restrict__ out) {
    long gid = (long)blockIdx.x * 256 + threadIdx.x;
    if (gid >= (long)MR * 64) return;
    int h = gid & 63;
    long m = gid >> 6;
    int n = (int)(m & (Nn - 1));
    long bt = m >> 10;
    int t = (int)(bt & (Tt - 1));
    int b = (int)(bt >> 4);
    out[gid] = f2bf(xg[(((long)b * Nn + n) * Hh + h) * Tt + t]);
}

// W_tc' [hin + 64*k][hout] = W_tc[hout][hin][k]   (f32)
__global__ void k_prep_wtc(const float* __restrict__ wtc, float* __restrict__ out) {
    int i = blockIdx.x * 256 + threadIdx.x;
    if (i >= 192 * 64) return;
    int row = i >> 6, col = i & 63;
    int hin = row & 63, kk = row >> 6;
    out[i] = wtc[((long)col * Hh + hin) * 3 + kk];
}

// Temporal window matrix: A[(bn*T+t)*192 + 64*kk + h] = x[bn,h,t+kk-1] + PE  (zero-pad)
__global__ void k_prep_atemp(const float* __restrict__ xg, __bf16* __restrict__ out) {
    long gid = (long)blockIdx.x * 256 + threadIdx.x;
    if (gid >= (long)BNT * 192) return;
    long m = gid / 192;
    int k = (int)(gid - m * 192);
    int kk = k >> 6, h = k & 63;
    long bn = m >> 4;
    int t = (int)(m & 15);
    int ts = t + kk - 1;
    float v = 0.f;
    if (ts >= 0 && ts < Tt) {
        float d = __expf(-(float)(h & ~1) * PE_SCALE);
        float ang = (float)ts * d;
        float pe = (h & 1) ? __cosf(ang) : __sinf(ang);
        v = xg[(bn * Hh + h) * Tt + ts] + pe;
    }
    out[gid] = f2bf(v);
}

// scatter: agg[bt,tgt,h] += lx[bt,src,h] * mw[e]
__global__ void k_scatter(const __bf16* __restrict__ lx, const float* __restrict__ mw,
                          const int* __restrict__ src, const int* __restrict__ tgt,
                          float* __restrict__ agg, int E, int NN, long total) {
    long gid = (long)blockIdx.x * 256 + threadIdx.x;
    if (gid >= total) return;
    int h = gid & 63;
    long r = gid >> 6;
    int e = (int)(r % E);
    int bt = (int)(r / E);
    float w = mw[e];
    float v = (float)lx[((long)bt * NN + src[e]) * 64 + h] * w;
    atomicAdd(&agg[((long)bt * NN + tgt[e]) * 64 + h], v);
}

// pooled_bf[(bt*N+n)*64+h] = mean_f cupd[((bt*N+n)*F+f)*64+h]
__global__ void k_pool(const float* __restrict__ cupd, __bf16* __restrict__ pooled) {
    long gid = (long)blockIdx.x * 256 + threadIdx.x;
    if (gid >= (long)MR * 64) return;
    int h = gid & 63;
    long m = gid >> 6;
    float s = 0.f;
    long base = m * (long)Ff * 64 + h;
#pragma unroll
    for (int f = 0; f < Ff; ++f) s += cupd[base + (long)f * 64];
    pooled[gid] = f2bf(s * 0.125f);
}

// LayerNorm over H=64 (wave per row) + transpose-store to pf_out[b,n,f,h,t]
__global__ void k_ln_pf(const float* __restrict__ cupd, const float* __restrict__ g,
                        const float* __restrict__ bt_, float* __restrict__ pf_out) {
    int lane = threadIdx.x & 31;
    int wave = threadIdx.x >> 5;
    long row = (long)blockIdx.x * 8 + wave;  // rows = MC
    const float* r = cupd + row * 64;
    int h0 = lane * 2;
    float v0 = r[h0], v1 = r[h0 + 1];
    float s = v0 + v1, q = v0 * v0 + v1 * v1;
#pragma unroll
    for (int m = 1; m < 32; m <<= 1) { s += __shfl_xor(s, m, 32); q += __shfl_xor(q, m, 32); }
    float mean = s * (1.f / 64.f);
    float inv = rsqrtf(q * (1.f / 64.f) - mean * mean + 1e-5f);
    // row = ((b*T + t)*N + n)*F + f
    int f = (int)(row & 7);
    int n = (int)((row >> 3) & (Nn - 1));
    int t = (int)((row >> 13) & (Tt - 1));
    int b = (int)(row >> 17);
    long obase = ((((long)b * Nn + n) * Ff + f) * Hh) * Tt + t;
    pf_out[obase + (long)h0 * Tt]       = (v0 - mean) * inv * g[h0]     + bt_[h0];
    pf_out[obase + (long)(h0 + 1) * Tt] = (v1 - mean) * inv * g[h0 + 1] + bt_[h0 + 1];
}

// LayerNorm + SiLU (wave per row), bf16 out [BNT,64]
__global__ void k_ln_silu(const float* __restrict__ fz, const float* __restrict__ g,
                          const float* __restrict__ bt_, __bf16* __restrict__ out) {
    int lane = threadIdx.x & 31;
    int wave = threadIdx.x >> 5;
    long row = (long)blockIdx.x * 8 + wave;  // rows = BNT
    const float* r = fz + row * 64;
    int h0 = lane * 2;
    float v0 = r[h0], v1 = r[h0 + 1];
    float s = v0 + v1, q = v0 * v0 + v1 * v1;
#pragma unroll
    for (int m = 1; m < 32; m <<= 1) { s += __shfl_xor(s, m, 32); q += __shfl_xor(q, m, 32); }
    float mean = s * (1.f / 64.f);
    float inv = rsqrtf(q * (1.f / 64.f) - mean * mean + 1e-5f);
    float y0 = (v0 - mean) * inv * g[h0] + bt_[h0];
    float y1 = (v1 - mean) * inv * g[h0 + 1] + bt_[h0 + 1];
    y0 = y0 / (1.f + __expf(-y0));
    y1 = y1 / (1.f + __expf(-y1));
    out[row * 64 + h0]     = f2bf(y0);
    out[row * 64 + h0 + 1] = f2bf(y1);
}

// x_spatial [b,t,n,h] (bf16) -> [b,n,t,h] (bf16)
__global__ void k_xsp_T(const __bf16* __restrict__ xsp, __bf16* __restrict__ out) {
    long gid = (long)blockIdx.x * 256 + threadIdx.x;
    if (gid >= (long)BNT * 64) return;
    int h = gid & 63;
    long m = gid >> 6;            // (b,n,t)
    int t = (int)(m & 15);
    long bn = m >> 4;
    int n = (int)(bn & (Nn - 1));
    int b = (int)(bn >> 10);
    out[gid] = xsp[((((long)b * Tt + t) * Nn + n) * 64) + h];
}

// ---------------------------------------------------------------------------
// Launcher
// ---------------------------------------------------------------------------
extern "C" void kernel_launch(void* const* d_in, const int* in_sizes, int n_in,
                              void* d_out, int out_size, void* d_ws, size_t ws_size,
                              hipStream_t stream) {
    (void)in_sizes; (void)n_in; (void)out_size; (void)ws_size;

    const float* x_global = (const float*)d_in[0];
    const float* pfx      = (const float*)d_in[1];
    const float* r_attr   = (const float*)d_in[2];
    const float* c_ew     = (const float*)d_in[3];
    const float* W_tc     = (const float*)d_in[4];
    const float* b_tc     = (const float*)d_in[5];
    const float* r_lin_W  = (const float*)d_in[6];
    const float* r_lin_b  = (const float*)d_in[7];
    const float* r_upd_W  = (const float*)d_in[8];
    const float* r_upd_b  = (const float*)d_in[9];
    const float* r_edge_W = (const float*)d_in[10];
    const float* r_edge_b = (const float*)d_in[11];
    const float* r_gate   = (const float*)d_in[12];
    const float* c_lin_W  = (const float*)d_in[13];
    const float* c_lin_b  = (const float*)d_in[14];
    const float* c_upd_W  = (const float*)d_in[15];
    const float* c_upd_b  = (const float*)d_in[16];
    const float* c_gate   = (const float*)d_in[17];
    const float* sf_W     = (const float*)d_in[18];
    const float* sf_b     = (const float*)d_in[19];
    const float* femb     = (const float*)d_in[20];
    const float* pfn_g    = (const float*)d_in[21];
    const float* pfn_b    = (const float*)d_in[22];
    const float* f1_W     = (const float*)d_in[23];
    const float* f1_b     = (const float*)d_in[24];
    const float* fln_g    = (const float*)d_in[25];
    const float* fln_b    = (const float*)d_in[26];
    const float* f2_W     = (const float*)d_in[27];
    const float* f2_b     = (const float*)d_in[28];
    const int*   r_idx    = (const int*)d_in[29];   // [2, ER]
    const int*   c_idx    = (const int*)d_in[30];   // [2, EC]

    float* x_out  = (float*)d_out;
    float* pf_out = (float*)d_out + XOUT_ELEMS;

    // workspace carving (256B aligned)
    char* base = (char*)d_ws;
    size_t off = 0;
    auto ALLOC = [&](size_t bytes) -> void* {
        void* r = base + off;
        off = (off + bytes + 255) & ~(size_t)255;
        return r;
    };
    float*  wtcp      = (float*) ALLOC((size_t)192 * 64 * 4);
    float*  mwr       = (float*) ALLOC((size_t)ER * 4);
    float*  mwc       = (float*) ALLOC((size_t)EC * 4);
    __bf16* nodes_bf  = (__bf16*)ALLOC((size_t)MC * 64 * 2);
    __bf16* clx_bf    = (__bf16*)ALLOC((size_t)MC * 64 * 2);
    float*  agg_c     = (float*) ALLOC((size_t)MC * 64 * 4);   // reused as cupd
    __bf16* aggc_bf   = (__bf16*)ALLOC((size_t)MC * 64 * 2);
    float*  cupd      = agg_c;                                  // alias (agg_c dead)
    __bf16* pooled_bf = (__bf16*)ALLOC((size_t)MR * 64 * 2);
    __bf16* xbt_bf    = (__bf16*)ALLOC((size_t)MR * 64 * 2);
    __bf16* lx_bf     = (__bf16*)ALLOC((size_t)MR * 64 * 2);
    float*  agg_r     = (float*) ALLOC((size_t)MR * 64 * 4);
    __bf16* aggr_bf   = (__bf16*)ALLOC((size_t)MR * 64 * 2);
    __bf16* rupd_bf   = (__bf16*)ALLOC((size_t)MR * 64 * 2);
    __bf16* xsp_bf    = (__bf16*)ALLOC((size_t)MR * 64 * 2);
    __bf16* xspT_bf   = (__bf16*)ALLOC((size_t)MR * 64 * 2);
    __bf16* atemp_bf  = (__bf16*)ALLOC((size_t)BNT * 192 * 2);
    __bf16* xtemp_bf  = (__bf16*)ALLOC((size_t)BNT * 64 * 2);
    float*  fz        = (float*) ALLOC((size_t)BNT * 64 * 4);
    __bf16* fzs_bf    = (__bf16*)ALLOC((size_t)BNT * 64 * 2);

    const dim3 T256(256);
    const long scat_c = (long)Bc * Tt * EC * 64;
    const long scat_r = (long)Bc * Tt * ER * 64;

    // --- edge weights / weight prep ---
    k_edge_w_river <<<(ER + 255) / 256, T256, 0, stream>>>(r_attr, r_edge_W, r_edge_b, r_gate, mwr);
    k_edge_w_causal<<<(EC + 255) / 256, T256, 0, stream>>>(c_ew, c_gate, mwc);
    k_prep_wtc     <<<(192 * 64 + 255) / 256, T256, 0, stream>>>(W_tc, wtcp);

    // --- causal graph conv ---
    k_prep_nodes<<<(unsigned)((long)MC * 64 / 256), T256, 0, stream>>>(pfx, femb, nodes_bf);
    gemm_wmma<64, 64, 0, 1><<<MC / 128, T256, 0, stream>>>(nodes_bf, 64, nodes_bf, 64, c_lin_W, c_lin_b, clx_bf);
    k_zero_f32<<<(unsigned)((long)MC * 64 / 256), T256, 0, stream>>>(agg_c, (long)MC * 64);
    k_scatter<<<(unsigned)(scat_c / 256), T256, 0, stream>>>(clx_bf, mwc, c_idx, c_idx + EC, agg_c, EC, NFc, scat_c);
    k_f32_to_bf16<<<(unsigned)((long)MC * 64 / 256), T256, 0, stream>>>(agg_c, aggc_bf, (long)MC * 64);
    gemm_wmma<128, 64, 1, 0><<<MC / 128, T256, 0, stream>>>(aggc_bf, 64, nodes_bf, 64, c_upd_W, c_upd_b, cupd);
    k_ln_pf<<<MC / 8, T256, 0, stream>>>(cupd, pfn_g, pfn_b, pf_out);
    k_pool <<<(unsigned)((long)MR * 64 / 256), T256, 0, stream>>>(cupd, pooled_bf);

    // --- river graph conv ---
    k_prep_xbt<<<(unsigned)((long)MR * 64 / 256), T256, 0, stream>>>(x_global, xbt_bf);
    gemm_wmma<64, 64, 0, 1><<<MR / 128, T256, 0, stream>>>(xbt_bf, 64, xbt_bf, 64, r_lin_W, r_lin_b, lx_bf);
    k_zero_f32<<<(unsigned)((long)MR * 64 / 256), T256, 0, stream>>>(agg_r, (long)MR * 64);
    k_scatter<<<(unsigned)(scat_r / 256), T256, 0, stream>>>(lx_bf, mwr, r_idx, r_idx + ER, agg_r, ER, Nn, scat_r);
    k_f32_to_bf16<<<(unsigned)((long)MR * 64 / 256), T256, 0, stream>>>(agg_r, aggr_bf, (long)MR * 64);
    gemm_wmma<128, 64, 1, 1><<<MR / 128, T256, 0, stream>>>(aggr_bf, 64, xbt_bf, 64, r_upd_W, r_upd_b, rupd_bf);

    // --- spatial fusion: silu([river_upd, pooled] @ sf_W + b) ---
    gemm_wmma<128, 64, 2, 1><<<MR / 128, T256, 0, stream>>>(rupd_bf, 64, pooled_bf, 64, sf_W, sf_b, xsp_bf);
    k_xsp_T<<<(unsigned)((long)BNT * 64 / 256), T256, 0, stream>>>(xsp_bf, xspT_bf);

    // --- temporal conv (as K=192 GEMM over padded windows + PE) ---
    k_prep_atemp<<<(unsigned)((long)BNT * 192 / 256), T256, 0, stream>>>(x_global, atemp_bf);
    gemm_wmma<192, 192, 0, 1><<<BNT / 128, T256, 0, stream>>>(atemp_bf, 192, atemp_bf, 192, wtcp, b_tc, xtemp_bf);

    // --- fusion head ---
    gemm_wmma<128, 64, 0, 0><<<BNT / 128, T256, 0, stream>>>(xtemp_bf, 64, xspT_bf, 64, f1_W, f1_b, fz);
    k_ln_silu<<<BNT / 8, T256, 0, stream>>>(fz, fln_g, fln_b, fzs_bf);
    gemm_wmma<64, 64, 0, 2><<<BNT / 128, T256, 0, stream>>>(fzs_bf, 64, fzs_bf, 64, f2_W, f2_b, x_out);
}
